// QuantumFeedForward_65481071409452
// MI455X (gfx1250) — compile-verified, matrix operand order checked by reading, not run on hardware
//
#include <hip/hip_runtime.h>
#include <stdint.h>

// ---------- types ----------
typedef __attribute__((ext_vector_type(16))) __bf16 v16bf;
typedef __attribute__((ext_vector_type(8)))  __bf16 v8bf;
typedef __attribute__((ext_vector_type(4)))  __bf16 v4bf;
typedef __attribute__((ext_vector_type(8)))  float  v8f;

// ---------- problem constants ----------
#define BB 4
#define SS 2048
#define EE 768
#define FF 3072
#define MM (BB * SS)   // 8192 rows

// ---------- elementwise: q = bf16(cos(x) * cos(theta)) ----------
__global__ __launch_bounds__(256) void qprep_kernel(
    const float* __restrict__ x, const float* __restrict__ p,
    __bf16* __restrict__ q, int total, int E) {
  int i = (blockIdx.x * blockDim.x + threadIdx.x) * 4;
  if (i >= total) return;
  int e = i % E;  // E % 4 == 0, so e..e+3 stay inside one row
  float4 xv = *(const float4*)&x[i];
  v4bf o;
  o[0] = (__bf16)(__cosf(xv.x) * __cosf(p[e + 0]));
  o[1] = (__bf16)(__cosf(xv.y) * __cosf(p[e + 1]));
  o[2] = (__bf16)(__cosf(xv.z) * __cosf(p[e + 2]));
  o[3] = (__bf16)(__cosf(xv.w) * __cosf(p[e + 3]));
  *(v4bf*)&q[i] = o;
}

// ---------- f32 -> bf16 weight conversion ----------
__global__ __launch_bounds__(256) void cvt_bf16_kernel(
    const float* __restrict__ src, __bf16* __restrict__ dst, int total) {
  int i = (blockIdx.x * blockDim.x + threadIdx.x) * 4;
  if (i >= total) return;
  float4 v = *(const float4*)&src[i];
  v4bf o;
  o[0] = (__bf16)v.x; o[1] = (__bf16)v.y; o[2] = (__bf16)v.z; o[3] = (__bf16)v.w;
  *(v4bf*)&dst[i] = o;
}

// ---------- WMMA GEMM: C[M,N] = act(A[M,K] * W[N,K]^T + bias) ----------
// A row-major bf16, W row-major [N,K] bf16, bias f32.
// 128x128 block tile, BK=64 (2 WMMA K-steps), 8 waves, double-buffered LDS
// filled with global_load_async_to_lds_b128 (ASYNCcnt pipeline).
template <bool RELU, typename OutT>
__global__ __launch_bounds__(256) void wmma_gemm_bias(
    const __bf16* __restrict__ A, const __bf16* __restrict__ W,
    const float* __restrict__ bias, OutT* __restrict__ C,
    int M, int N, int K) {
  constexpr int BM = 128, BN = 128, BK = 64;
  __shared__ __align__(16) __bf16 As[2][BM * BK];  // 2 x 16 KB
  __shared__ __align__(16) __bf16 Bs[2][BN * BK];  // 2 x 16 KB

  const int tid  = threadIdx.x;
  const int lane = tid & 31;
  const int wid  = tid >> 5;     // 8 waves
  const int wm   = wid & 3;      // wave row slab: 4 x 32 rows
  const int wn   = wid >> 2;     // wave col slab: 2 x 64 cols
  const int half = lane >> 4;    // lane half selects K-run (ISA 16-bit layout)
  const int lsub = lane & 15;

  const int m0 = blockIdx.y * BM;
  const int n0 = blockIdx.x * BN;

  __builtin_prefetch(&bias[n0], 0, 0);  // global_prefetch_b8

  // ---- per-thread staging chunks: 1024 x 16B per tile -> 4 per thread ----
  uint64_t gA[4], gB[4];
  uint32_t lA0[4], lA1[4], lB0[4], lB1[4];
#pragma unroll
  for (int i = 0; i < 4; ++i) {
    int c   = tid * 4 + i;       // 0..1023
    int row = c >> 3;            // 8 chunks per 64-wide row
    int cc  = (c & 7) << 3;      // element offset 0..56
    gA[i]  = (uint64_t)(uintptr_t)(A + (size_t)(m0 + row) * K + cc);
    gB[i]  = (uint64_t)(uintptr_t)(W + (size_t)(n0 + row) * K + cc);
    lA0[i] = (uint32_t)(size_t)&As[0][row * BK + cc];
    lA1[i] = (uint32_t)(size_t)&As[1][row * BK + cc];
    lB0[i] = (uint32_t)(size_t)&Bs[0][row * BK + cc];
    lB1[i] = (uint32_t)(size_t)&Bs[1][row * BK + cc];
  }

  // issue one tile's async copies (8 instructions/thread, ASYNCcnt += 8)
  auto stage = [&](int kt, const uint32_t* la, const uint32_t* lb) {
    size_t koff = (size_t)kt * (BK * 2);  // byte offset along K
#pragma unroll
    for (int i = 0; i < 4; ++i) {
      asm volatile("global_load_async_to_lds_b128 %0, %1, off"
                   :: "v"(la[i]), "v"(gA[i] + koff) : "memory");
      asm volatile("global_load_async_to_lds_b128 %0, %1, off"
                   :: "v"(lb[i]), "v"(gB[i] + koff) : "memory");
    }
  };

  v8f acc[2][4] = {};

  // consume one LDS tile: 2 K-steps x (2x4 subtiles) = 16 WMMA
  auto compute = [&](const __bf16* as, const __bf16* bs) {
#pragma unroll
    for (int ks = 0; ks < 2; ++ks) {
      const int kb = ks * 32 + half * 8;
      v16bf af[2];
#pragma unroll
      for (int sm = 0; sm < 2; ++sm) {
        int r = wm * 32 + sm * 16 + lsub;
        v8bf lo = *(const v8bf*)&as[r * BK + kb];
        v8bf hi = *(const v8bf*)&as[r * BK + kb + 16];
#pragma unroll
        for (int j = 0; j < 8; ++j) { af[sm][j] = lo[j]; af[sm][j + 8] = hi[j]; }
      }
#pragma unroll
      for (int sn = 0; sn < 4; ++sn) {
        int n = wn * 64 + sn * 16 + lsub;
        v8bf lo = *(const v8bf*)&bs[n * BK + kb];
        v8bf hi = *(const v8bf*)&bs[n * BK + kb + 16];
        v16bf bf;
#pragma unroll
        for (int j = 0; j < 8; ++j) { bf[j] = lo[j]; bf[j + 8] = hi[j]; }
#pragma unroll
        for (int sm = 0; sm < 2; ++sm) {
          acc[sm][sn] = __builtin_amdgcn_wmma_f32_16x16x32_bf16(
              false, af[sm], false, bf, (short)0, acc[sm][sn], false, false);
        }
      }
    }
  };

  const int nt = K / BK;  // 12 or 48 — always even here
  stage(0, lA0, lB0);

  for (int t = 0; t < nt; t += 2) {
    // tile t is (or will be) in buffer 0; prefetch t+1 into buffer 1
    stage(t + 1, lA1, lB1);
    asm volatile("s_wait_asynccnt 0x8" ::: "memory");  // tile t landed
    __syncthreads();
    compute(As[0], Bs[0]);
    __syncthreads();                                   // buffer 0 free

    if (t + 2 < nt) {
      stage(t + 2, lA0, lB0);
      asm volatile("s_wait_asynccnt 0x8" ::: "memory");  // tile t+1 landed
    } else {
      asm volatile("s_wait_asynccnt 0x0" ::: "memory");
    }
    __syncthreads();
    compute(As[1], Bs[1]);
    __syncthreads();                                   // buffer 1 free
  }

  // ---- epilogue: bias (+ReLU); C/D layout: M = r + 8*half, N = lsub ----
#pragma unroll
  for (int sm = 0; sm < 2; ++sm) {
#pragma unroll
    for (int sn = 0; sn < 4; ++sn) {
      int col = n0 + wn * 64 + sn * 16 + lsub;
      float bv = bias[col];
#pragma unroll
      for (int r = 0; r < 8; ++r) {
        int row = m0 + wm * 32 + sm * 16 + half * 8 + r;
        float v = acc[sm][sn][r] + bv;
        if (RELU) v = v > 0.0f ? v : 0.0f;
        C[(size_t)row * N + col] = (OutT)v;
      }
    }
  }
}

// ---------- launch ----------
extern "C" void kernel_launch(void* const* d_in, const int* in_sizes, int n_in,
                              void* d_out, int out_size, void* d_ws, size_t ws_size,
                              hipStream_t stream) {
  const float* x  = (const float*)d_in[0];
  const float* th = (const float*)d_in[1];
  const float* W1 = (const float*)d_in[2];
  const float* b1 = (const float*)d_in[3];
  const float* W2 = (const float*)d_in[4];
  const float* b2 = (const float*)d_in[5];
  float* out = (float*)d_out;

  // workspace layout (bytes)
  const size_t q_bytes  = (size_t)MM * EE * 2;  // 12.6 MB
  const size_t w1_bytes = (size_t)FF * EE * 2;  // 4.7 MB
  const size_t w2_bytes = (size_t)EE * FF * 2;  // 4.7 MB
  char* ws = (char*)d_ws;
  __bf16* qb  = (__bf16*)ws;
  __bf16* w1b = (__bf16*)(ws + q_bytes);
  __bf16* w2b = (__bf16*)(ws + q_bytes + w1_bytes);
  __bf16* hb  = (__bf16*)(ws + q_bytes + w1_bytes + w2_bytes);  // 50 MB

  const int qn = MM * EE;      // 6,291,456
  const int wn = FF * EE;      // 2,359,296

  qprep_kernel<<<(qn / 4 + 255) / 256, 256, 0, stream>>>(x, th, qb, qn, EE);
  cvt_bf16_kernel<<<(wn / 4 + 255) / 256, 256, 0, stream>>>(W1, w1b, wn);
  cvt_bf16_kernel<<<(wn / 4 + 255) / 256, 256, 0, stream>>>(W2, w2b, wn);

  // GEMM1: h = relu(q @ W1^T + b1)  -> bf16 [8192, 3072]
  wmma_gemm_bias<true, __bf16>
      <<<dim3(FF / 128, MM / 128), 256, 0, stream>>>(qb, w1b, b1, hb, MM, FF, EE);
  // GEMM2: out = h @ W2^T + b2      -> f32 [8192, 768]
  wmma_gemm_bias<false, float>
      <<<dim3(EE / 128, MM / 128), 256, 0, stream>>>(hb, w2b, b2, out, MM, EE, FF);
}
